// AttentionHead_88012469829751
// MI455X (gfx1250) — matrix operand order repaired and checked by reading, not compiled, not run on hardware
//
#include <hip/hip_runtime.h>
#include <math.h>

#define BATCH  8
#define SEQ    2048
#define DMODEL 2048
#define HDIM   128

typedef __attribute__((ext_vector_type(16))) __bf16 v16bf;
typedef __attribute__((ext_vector_type(8)))  __bf16 v8bf;
typedef __attribute__((ext_vector_type(4)))  __bf16 v4bf;
typedef __attribute__((ext_vector_type(8)))  float  v8f;
typedef __attribute__((ext_vector_type(4)))  unsigned int v4u;
typedef __attribute__((ext_vector_type(8)))  int v8i;
typedef __attribute__((ext_vector_type(4)))  int v4i;

static __device__ __forceinline__ v8f wmma_bf16(v16bf a, v16bf b, v8f c) {
  return __builtin_amdgcn_wmma_f32_16x16x32_bf16(false, a, false, b, (short)0, c,
                                                 false, false);
}

static __device__ __forceinline__ v16bf mk_frag(v8bf lo, v8bf hi) {
  return __builtin_shufflevector(lo, hi, 0,1,2,3,4,5,6,7,8,9,10,11,12,13,14,15);
}

// Global-memory 16-bit WMMA A/B fragment (16 rows x 32 K), K contiguous,
// row stride `stride` elements. CDNA5 layout: lanes 0-15 rows 0-15 K{0..7,16..23},
// lanes 16-31 rows 0-15 K{8..15,24..31}.
static __device__ __forceinline__ v16bf load_frag(const __bf16* base, int stride, int lane) {
  const int row = lane & 15;
  const int kh  = (lane >> 4) << 3;
  const __bf16* p = base + (size_t)row * stride + kh;
  return mk_frag(*(const v8bf*)(p), *(const v8bf*)(p + 16));
}

// ---- Tensor Data Mover: 2D tile load Global -> LDS with row padding ----
// D# packing per CDNA5 ISA 8.3/8.4. data_size = 2 bytes (bf16).
// padded LDS: +16B after every (1<<(pad_ic+1)) DWORDs of payload.
static __device__ __forceinline__ void tdm_load_2d(unsigned int lds_byte,
                                                   const void* gaddr,
                                                   unsigned int dim0, unsigned int dim1,
                                                   unsigned int tile0, unsigned int tile1,
                                                   unsigned long long stride0_elems,
                                                   unsigned int pad_ic, unsigned int pad_ac) {
  unsigned long long ga = (unsigned long long)(uintptr_t)gaddr;
  v4u g0;
  g0[0] = 1u;                                             // count=1 user descriptor
  g0[1] = lds_byte;                                       // lds_addr [63:32]
  g0[2] = (unsigned int)(ga & 0xFFFFFFFFu);               // global_addr [95:64]
  g0[3] = (unsigned int)((ga >> 32) & 0x1FFFFFFu) | (2u << 30);  // addr[56:32] | type=2
  v8i g1;
  unsigned int w0 = (1u << 16)                            // data_size=1 -> 2 bytes
                  | (1u << 20)                            // pad_enable
                  | (pad_ic << 22) | (pad_ac << 25);
  g1[0] = (int)w0;
  g1[1] = (int)((dim0 & 0xFFFFu) << 16);                  // tensor_dim0[15:0] @ [63:48]
  g1[2] = (int)(((dim0 >> 16) & 0xFFFFu) | ((dim1 & 0xFFFFu) << 16));
  g1[3] = (int)(((dim1 >> 16) & 0xFFFFu) | ((tile0 & 0xFFFFu) << 16));
  g1[4] = (int)(tile1 & 0xFFFFu);                         // tile_dim1; tile_dim2=0
  g1[5] = (int)(unsigned int)(stride0_elems & 0xFFFFFFFFu);
  g1[6] = (int)(unsigned int)((stride0_elems >> 32) & 0xFFFFu);
  g1[7] = 0;
  v4i z4 = {0, 0, 0, 0};
  v8i z8 = {0, 0, 0, 0, 0, 0, 0, 0};
  // 6-arg toolchain variant: (g0, g1, g2, g3, g4, cpol)
  __builtin_amdgcn_tensor_load_to_lds(g0, g1, z4, z4, z8, 0);
}

// ---------------- conversion kernels ----------------

__global__ void __launch_bounds__(256) cvt_x_kernel(const float* __restrict__ x,
                                                    __bf16* __restrict__ xb, int n4) {
  int i = blockIdx.x * 256 + threadIdx.x;
  if (i >= n4) return;
  float4 v = ((const float4*)x)[i];
  v4bf o;
  o[0] = (__bf16)v.x; o[1] = (__bf16)v.y; o[2] = (__bf16)v.z; o[3] = (__bf16)v.w;
  ((v4bf*)xb)[i] = o;
}

// Wt[kind][n][k] = W[kind][k][n]
__global__ void __launch_bounds__(256) cvt_w_kernel(const float* __restrict__ Wq,
                                                    const float* __restrict__ Wk,
                                                    const float* __restrict__ Wv,
                                                    __bf16* __restrict__ Wtb) {
  int i = blockIdx.x * 256 + threadIdx.x;
  int kind = i / (DMODEL * HDIM);
  int rem  = i % (DMODEL * HDIM);
  int k = rem / HDIM, n = rem % HDIM;
  const float* W = (kind == 0) ? Wq : (kind == 1 ? Wk : Wv);
  Wtb[(size_t)kind * HDIM * DMODEL + (size_t)n * DMODEL + k] =
      (__bf16)W[(size_t)k * HDIM + n];
}

// ---------------- QKV projection GEMM ----------------
// A tile (16 x 2048 bf16, 64KB) staged once per block by the TDM with 16B per 1KB
// padding (row stride 4160B -> lanes land on distinct banks). 8 waves, one
// 16-wide n-tile each; B fragments streamed from L2-resident W^T.

__global__ void __launch_bounds__(256) qkv_gemm_kernel(const __bf16* __restrict__ xb,
                                                       const __bf16* __restrict__ Wtb,
                                                       __bf16* __restrict__ Qb,
                                                       __bf16* __restrict__ Kb,
                                                       __bf16* __restrict__ Vtb) {
  __shared__ __align__(16) __bf16 Alds[33280];   // 16 * (4096 + 64) bytes
  const int lane = threadIdx.x & 31;
  const int wave = threadIdx.x >> 5;
  const int m0   = blockIdx.x * 16;
  const int kind = blockIdx.y;

  if (wave == 0) {
    // 16 rows x 2048 cols, pad 4 DWORDs per 256 DWORDs (codes 7,3)
    tdm_load_2d((unsigned int)(uintptr_t)&Alds[0], xb + (size_t)m0 * DMODEL,
                DMODEL, 16, DMODEL, 16, DMODEL, 7u, 3u);
    __builtin_amdgcn_s_wait_tensorcnt(0);
  }
  __syncthreads();

  const __bf16* Wt = Wtb + (size_t)kind * HDIM * DMODEL + (size_t)(wave * 16) * DMODEL;
  const char* ab = (const char*)&Alds[0];
  const int row = lane & 15;
  const int kh  = (lane >> 4) << 3;

  v8f c = {};
  for (int k = 0; k < DMODEL; k += 32) {
    __builtin_prefetch(Wt + k + 256, 0, 0);
    unsigned int c0 = (unsigned int)(k + kh) * 2u;
    unsigned int c1 = c0 + 32u;
    v8bf lo = *(const v8bf*)(ab + row * 4160 + c0 + ((c0 >> 10) << 4));
    v8bf hi = *(const v8bf*)(ab + row * 4160 + c1 + ((c1 >> 10) << 4));
    v16bf a = mk_frag(lo, hi);
    v16bf b = load_frag(Wt + k, DMODEL, lane);
    c = wmma_bf16(a, b, c);
  }

  const int h = wave * 16 + (lane & 15);
  const float scale = (kind == 0) ? 0.08838834764831845f : 1.0f;  // 1/sqrt(128)
#pragma unroll
  for (int r = 0; r < 8; ++r) {
    int gr = m0 + r + ((lane >> 4) << 3);
    float v = c[r] * scale;
    if (kind == 0) {
      Qb[(size_t)gr * HDIM + h] = (__bf16)v;
    } else if (kind == 1) {
      Kb[(size_t)gr * HDIM + h] = (__bf16)v;
    } else {
      int bb = gr / SEQ, s = gr % SEQ;
      Vtb[((size_t)bb * HDIM + h) * SEQ + s] = (__bf16)v;
    }
  }
}

// ---------------- flash attention ----------------
// 4 waves = 4 consecutive 16-query tiles of one batch. K (32x128) and V^T
// (128x32) key-blocks double-buffered in LDS via TDM (padded rows: 272B / 80B),
// DMA of block n+1 overlapped with WMMA compute of block n.

__global__ void __launch_bounds__(128) attn_kernel(const __bf16* __restrict__ Qb,
                                                   const __bf16* __restrict__ Kb,
                                                   const __bf16* __restrict__ Vtb,
                                                   float* __restrict__ out) {
  __shared__ __align__(16) __bf16 Kls[2][4352];   // 32 * 272B
  __shared__ __align__(16) __bf16 Vls[2][5120];   // 128 * 80B
  __shared__ __align__(16) __bf16 plds[4][512];   // per-wave P staging
  const int lane = threadIdx.x & 31;
  const int wid  = threadIdx.x >> 5;
  const int b     = blockIdx.x >> 5;              // 32 blocks per batch
  const int qbase = (blockIdx.x & 31) << 6;
  const int q0    = qbase + wid * 16;
  const size_t bS = (size_t)b * SEQ;
  const int kh = (lane >> 4) << 3;

  // Q fragments (H=128 -> 4 chunks of K=32)
  v16bf qf[4];
#pragma unroll
  for (int kf = 0; kf < 4; ++kf)
    qf[kf] = load_frag(Qb + (bS + q0) * HDIM + kf * 32, HDIM, lane);

  v8f acc[8] = {};
  float mrow[8], lrow[8];
#pragma unroll
  for (int r = 0; r < 8; ++r) { mrow[r] = -__builtin_inff(); lrow[r] = 0.0f; }

  const int my_nkb  = ((q0 + 15) >> 5) + 1;
  const int nkb_max = (qbase >> 5) + 2;           // causal extent of the block

  auto issue_kv = [&](int kb2, int p) {
    // K block: 32 key-rows x 128 h, row stride H; pad 256B@16B -> 272B rows
    tdm_load_2d((unsigned int)(uintptr_t)&Kls[p][0], Kb + (bS + kb2) * HDIM,
                HDIM, 32, HDIM, 32, HDIM, 5u, 3u);
    // V block: 128 h-rows x 32 keys from Vt[b][h][s], row stride S; 64B@16B pad -> 80B rows
    tdm_load_2d((unsigned int)(uintptr_t)&Vls[p][0],
                Vtb + (size_t)b * HDIM * SEQ + kb2,
                32, HDIM, 32, HDIM, SEQ, 3u, 3u);
  };

  int par = 0;
  if (wid == 0) issue_kv(0, 0);

  for (int blk = 0; blk < nkb_max; ++blk) {
    const int kb = blk << 5;
    if (wid == 0) __builtin_amdgcn_s_wait_tensorcnt(0);
    __syncthreads();                               // buf[par] ready; prev compute done
    if (wid == 0 && blk + 1 < nkb_max) issue_kv((blk + 1) << 5, par ^ 1);

    if (blk < my_nkb) {
      const char* kbytes = (const char*)&Kls[par][0];
      const char* vbytes = (const char*)&Vls[par][0];

      // scores: two 16x16 tiles over this 32-key block
      v8f s0 = {}, s1 = {};
#pragma unroll
      for (int kf = 0; kf < 4; ++kf) {
        const char* p0 = kbytes + (lane & 15) * 272 + (kf * 32 + kh) * 2;
        const char* p1 = p0 + 16 * 272;
        v16bf k0 = mk_frag(*(const v8bf*)p0, *(const v8bf*)(p0 + 32));
        v16bf k1 = mk_frag(*(const v8bf*)p1, *(const v8bf*)(p1 + 32));
        s0 = wmma_bf16(qf[kf], k0, s0);
        s1 = wmma_bf16(qf[kf], k1, s1);
      }

      // causal mask on diagonal block(s)
      if (kb + 31 > q0) {
        const int key0 = kb + (lane & 15);
#pragma unroll
        for (int r = 0; r < 8; ++r) {
          int q = q0 + r + kh;
          if (key0 > q)      s0[r] = -__builtin_inff();
          if (key0 + 16 > q) s1[r] = -__builtin_inff();
        }
      }

      // online softmax (row spans a 16-lane half; xor masks 1..8 reduce it)
      float alpha[8];
#pragma unroll
      for (int r = 0; r < 8; ++r) {
        float bm = fmaxf(s0[r], s1[r]);
#pragma unroll
        for (int off = 8; off >= 1; off >>= 1) bm = fmaxf(bm, __shfl_xor(bm, off));
        float mnew = fmaxf(mrow[r], bm);
        alpha[r] = __expf(mrow[r] - mnew);
        float p0 = __expf(s0[r] - mnew);
        float p1 = __expf(s1[r] - mnew);
        s0[r] = p0; s1[r] = p1;
        float rs = p0 + p1;
#pragma unroll
        for (int off = 8; off >= 1; off >>= 1) rs += __shfl_xor(rs, off);
        lrow[r] = lrow[r] * alpha[r] + rs;
        mrow[r] = mnew;
      }
#pragma unroll
      for (int t = 0; t < 8; ++t)
#pragma unroll
        for (int r = 0; r < 8; ++r) acc[t][r] *= alpha[r];

      // P: C-layout -> row-major 16x32 bf16 in LDS -> A fragment
      {
        __bf16* pl = &plds[wid][0];
        const int col = lane & 15;
#pragma unroll
        for (int r = 0; r < 8; ++r) {
          pl[(r + kh) * 32 + col]      = (__bf16)s0[r];
          pl[(r + kh) * 32 + col + 16] = (__bf16)s1[r];
        }
      }
      const __bf16* pp = &plds[wid][0] + (lane & 15) * 32 + kh;
      v16bf pa = mk_frag(*(const v8bf*)pp, *(const v8bf*)(pp + 16));

      // P(16x32) x V(32x16) per h-tile, B fragments from padded V LDS rows
#pragma unroll
      for (int t = 0; t < 8; ++t) {
        const char* pv = vbytes + (t * 16 + (lane & 15)) * 80 + kh * 2;
        v16bf vf = mk_frag(*(const v8bf*)pv, *(const v8bf*)(pv + 32));
        acc[t] = wmma_bf16(pa, vf, acc[t]);
      }
    }

    __syncthreads();                               // done reading buf[par]
    par ^= 1;
  }

  // epilogue: normalize, store fp32
#pragma unroll
  for (int r = 0; r < 8; ++r) {
    float inv = 1.0f / lrow[r];
    int q = q0 + r + kh;
#pragma unroll
    for (int t = 0; t < 8; ++t)
      out[(bS + q) * HDIM + t * 16 + (lane & 15)] = acc[t][r] * inv;
  }
}

// ---------------- launcher ----------------

extern "C" void kernel_launch(void* const* d_in, const int* in_sizes, int n_in,
                              void* d_out, int out_size, void* d_ws, size_t ws_size,
                              hipStream_t stream) {
  const float* x  = (const float*)d_in[0];
  const float* Wq = (const float*)d_in[1];
  const float* Wk = (const float*)d_in[2];
  const float* Wv = (const float*)d_in[3];
  float* out = (float*)d_out;

  char* ws = (char*)d_ws;
  __bf16* xb  = (__bf16*)(ws);                               // 64 MB
  __bf16* Wtb = (__bf16*)(ws + (size_t)67108864);            // 1.5 MB
  __bf16* Qb  = (__bf16*)(ws + (size_t)68681728);            // 4 MB
  __bf16* Kb  = (__bf16*)(ws + (size_t)72876032);            // 4 MB
  __bf16* Vtb = (__bf16*)(ws + (size_t)77070336);            // 4 MB (transposed)

  const int n4 = BATCH * SEQ * DMODEL / 4;
  hipLaunchKernelGGL(cvt_x_kernel, dim3(n4 / 256), dim3(256), 0, stream, x, xb, n4);
  hipLaunchKernelGGL(cvt_w_kernel, dim3(3 * DMODEL * HDIM / 256), dim3(256), 0, stream,
                     Wq, Wk, Wv, Wtb);
  hipLaunchKernelGGL(qkv_gemm_kernel, dim3(BATCH * SEQ / 16, 3), dim3(256), 0, stream,
                     xb, Wtb, Qb, Kb, Vtb);
  hipLaunchKernelGGL(attn_kernel, dim3(BATCH * SEQ / 16 / 4), dim3(128), 0, stream,
                     Qb, Kb, Vtb, out);
  (void)in_sizes; (void)n_in; (void)out_size; (void)ws_size;
}
